// DynamicChannelProcessing_90915867722025
// MI455X (gfx1250) — matrix operand order repaired and checked by reading, not compiled, not run on hardware
//
#include <hip/hip_runtime.h>
#include <hip/hip_bf16.h>

#define DIMC   768
#define HEADS  8
#define CHD    96
#define HIDDEN 3072
#define BB     16
#define HS     32
#define WSZ    32
#define NTOK   1024
#define LN_EPS 1e-5f

typedef __attribute__((ext_vector_type(16))) __bf16 v16bf;
typedef __attribute__((ext_vector_type(8)))  __bf16 v8bf;
typedef __attribute__((ext_vector_type(8)))  float  v8f;

#if __has_builtin(__builtin_amdgcn_global_load_async_to_lds_b128) && \
    __has_builtin(__builtin_amdgcn_s_wait_asynccnt)
#define USE_ASYNC_LDS 1
typedef __attribute__((ext_vector_type(4))) int v4i;
typedef v4i __attribute__((address_space(1)))* gptr_v4i;   // global v4i*
typedef v4i __attribute__((address_space(3)))* lptr_v4i;   // LDS v4i*
#else
#define USE_ASYNC_LDS 0
#endif

static __device__ __forceinline__ unsigned short f2bf(float f) {
    union { float f; unsigned u; } v; v.f = f;
    unsigned u = v.u;
    u += 0x7FFFu + ((u >> 16) & 1u);   // round-to-nearest-even
    return (unsigned short)(u >> 16);
}
static __device__ __forceinline__ float bf2f(unsigned short h) {
    union { unsigned u; float f; } v; v.u = ((unsigned)h) << 16; return v.f;
}
static __device__ __forceinline__ float wred_sum(float v) {
    #pragma unroll
    for (int o = 16; o > 0; o >>= 1) v += __shfl_xor(v, o, 32);
    return v;
}
static __device__ __forceinline__ float wred_max(float v) {
    #pragma unroll
    for (int o = 16; o > 0; o >>= 1) v = fmaxf(v, __shfl_xor(v, o, 32));
    return v;
}

// ---------------------------------------------------------------- f32 -> bf16
__global__ __launch_bounds__(256) void cvt_f32_bf16(
    const float* __restrict__ in, unsigned short* __restrict__ out, int n)
{
    int i = blockIdx.x * 256 + threadIdx.x;
    if (i < n) out[i] = f2bf(in[i]);
}

// ---------------------------------------------------------------- WMMA GEMM
// C[m,n] = sum_k A[m,k] * W[n,k]  (+bias[n]); A,W bf16 row-major, f32 accum.
// Block tile 128x128, K-step 32, 8 waves (wave tile 32x64 = 2x4 WMMA frags).
// Staging uses GLOBAL_LOAD_ASYNC_TO_LDS_B128 (ASYNCcnt) when available.
__global__ __launch_bounds__(256) void gemm_bf16_wmma(
    const unsigned short* __restrict__ A,
    const unsigned short* __restrict__ Wt,
    const float* __restrict__ bias,
    float* __restrict__ outF,
    unsigned short* __restrict__ outB,
    int M, int Nout, int K)
{
    __shared__ unsigned short As[2][128][40];   // +8 halves pad
    __shared__ unsigned short Bs[2][128][40];

    const int tid  = threadIdx.x;
    const int lane = tid & 31;
    const int wid  = tid >> 5;
    const int wm   = wid & 3;       // wave M position (0..3)
    const int wn   = wid >> 2;      // wave N position (0..1)
    const int g    = lane >> 4;     // lane group
    const int lr   = lane & 15;

    const int n0 = blockIdx.x * 128;
    const int m0 = blockIdx.y * 128;

    const int ldRow = tid >> 1;         // 128 rows, 2 threads per row
    const int ldSeg = (tid & 1) * 16;   // 16 halves (32B) per thread

    v8f acc[2][4];
    #pragma unroll
    for (int i = 0; i < 2; ++i)
        #pragma unroll
        for (int j = 0; j < 4; ++j)
            #pragma unroll
            for (int r = 0; r < 8; ++r) acc[i][j][r] = 0.f;

    const int KT = K >> 5;

    auto stage = [&](int buf, int kt) {
        const unsigned short* pa = A  + (size_t)(m0 + ldRow) * K + kt * 32 + ldSeg;
        const unsigned short* pw = Wt + (size_t)(n0 + ldRow) * K + kt * 32 + ldSeg;
#if USE_ASYNC_LDS
        __builtin_amdgcn_global_load_async_to_lds_b128(
            (gptr_v4i)pa,       (lptr_v4i)&As[buf][ldRow][ldSeg],     0, 0);
        __builtin_amdgcn_global_load_async_to_lds_b128(
            (gptr_v4i)(pa + 8), (lptr_v4i)&As[buf][ldRow][ldSeg + 8], 0, 0);
        __builtin_amdgcn_global_load_async_to_lds_b128(
            (gptr_v4i)pw,       (lptr_v4i)&Bs[buf][ldRow][ldSeg],     0, 0);
        __builtin_amdgcn_global_load_async_to_lds_b128(
            (gptr_v4i)(pw + 8), (lptr_v4i)&Bs[buf][ldRow][ldSeg + 8], 0, 0);
#else
        uint4 a0 = ((const uint4*)pa)[0];
        uint4 a1 = ((const uint4*)pa)[1];
        uint4 w0 = ((const uint4*)pw)[0];
        uint4 w1 = ((const uint4*)pw)[1];
        *(uint4*)&As[buf][ldRow][ldSeg]     = a0;
        *(uint4*)&As[buf][ldRow][ldSeg + 8] = a1;
        *(uint4*)&Bs[buf][ldRow][ldSeg]     = w0;
        *(uint4*)&Bs[buf][ldRow][ldSeg + 8] = w1;
#endif
    };

    stage(0, 0);

    for (int kt = 0; kt < KT; ++kt) {
#if USE_ASYNC_LDS
        __builtin_amdgcn_s_wait_asynccnt(0);   // staged tile landed in LDS
#endif
        __syncthreads();
        if (kt + 1 < KT) stage((kt + 1) & 1, kt + 1);
        if (kt + 2 < KT) {  // pull the tile after next toward L2/WGP$
            __builtin_prefetch(A  + (size_t)(m0 + ldRow) * K + (kt + 2) * 32 + ldSeg, 0, 1);
            __builtin_prefetch(Wt + (size_t)(n0 + ldRow) * K + (kt + 2) * 32 + ldSeg, 0, 1);
        }
        const int buf = kt & 1;

        // Preload ALL fragments, then issue the 8 WMMAs back-to-back.
        v16bf af[2];
        #pragma unroll
        for (int fi = 0; fi < 2; ++fi) {
            const int row = wm * 32 + fi * 16 + lr;   // A: lane row, K per ISA layout
            v8bf lo = *(const v8bf*)&As[buf][row][g * 8];
            v8bf hi = *(const v8bf*)&As[buf][row][16 + g * 8];
            af[fi] = __builtin_shufflevector(lo, hi,
                0,1,2,3,4,5,6,7,8,9,10,11,12,13,14,15);
        }
        v16bf bfr[4];
        #pragma unroll
        for (int fj = 0; fj < 4; ++fj) {
            const int col = wn * 64 + fj * 16 + lr;   // B: N=lane%16, K per ISA layout (same as A)
            v8bf lo = *(const v8bf*)&Bs[buf][col][g * 8];
            v8bf hi = *(const v8bf*)&Bs[buf][col][16 + g * 8];
            bfr[fj] = __builtin_shufflevector(lo, hi,
                0,1,2,3,4,5,6,7,8,9,10,11,12,13,14,15);
        }
        #pragma unroll
        for (int fj = 0; fj < 4; ++fj)
            #pragma unroll
            for (int fi = 0; fi < 2; ++fi)
                acc[fi][fj] = __builtin_amdgcn_wmma_f32_16x16x32_bf16(
                    false, af[fi], false, bfr[fj], (short)0, acc[fi][fj], false, false);
    }
    __syncthreads();

    // Epilogue: C/D layout — VGPR r, lanes 0-15: M=r; lanes 16-31: M=r+8; N=lane%16
    #pragma unroll
    for (int fj = 0; fj < 4; ++fj) {
        const int n = n0 + wn * 64 + fj * 16 + lr;
        const float bv = bias ? bias[n] : 0.f;
        #pragma unroll
        for (int fi = 0; fi < 2; ++fi) {
            #pragma unroll
            for (int r = 0; r < 8; ++r) {
                const int m = m0 + wm * 32 + fi * 16 + g * 8 + r;
                float vv = acc[fi][fj][r] + bv;
                if (outF) outF[(size_t)m * Nout + n] = vv;
                else      outB[(size_t)m * Nout + n] = f2bf(vv);
            }
        }
    }
}

// ------------------------------------------------- per-column softmax stats
// One wave per column (b,c): max_n P[b,n,c], sum_n exp(P-max).
__global__ __launch_bounds__(256) void col_stats(
    const float* __restrict__ P, float* __restrict__ mx, float* __restrict__ se,
    int Nn, int C)
{
    const int lane = threadIdx.x & 31;
    const int wid  = threadIdx.x >> 5;
    const int col  = blockIdx.x * 8 + wid;
    const int b = col / C, c = col % C;
    const float* p = P + (size_t)b * Nn * C + c;

    float m = -3.0e38f;
    for (int n = lane; n < Nn; n += 32) m = fmaxf(m, p[(size_t)n * C]);
    m = wred_max(m);
    float s = 0.f;
    for (int n = lane; n < Nn; n += 32) s += __expf(p[(size_t)n * C] - m);
    s = wred_sum(s);
    if (lane == 0) { mx[col] = m; se[col] = s; }
}

// --------------------------------------- kp[b,h,n] = mean_ch softmax_n(x)
__global__ __launch_bounds__(256) void kp_kernel(
    const float* __restrict__ x, const float* __restrict__ mx,
    const float* __restrict__ se, float* __restrict__ kp)
{
    __shared__ float e[DIMC];
    const int lane = threadIdx.x & 31;
    const int wid  = threadIdx.x >> 5;       // wave == head
    const int bn = blockIdx.x;               // b*NTOK + n
    const int b = bn >> 10, n = bn & 1023;
    for (int c = threadIdx.x; c < DIMC; c += 256) {
        const int sc = b * DIMC + c;
        e[c] = __expf(x[(size_t)bn * DIMC + c] - mx[sc]) / se[sc];
    }
    __syncthreads();
    float s = 0.f;
    #pragma unroll
    for (int ch = lane; ch < CHD; ch += 32) s += e[wid * CHD + ch];
    s = wred_sum(s);
    if (lane == 0)
        kp[(size_t)(b * HEADS + wid) * NTOK + n] = s * (1.f / CHD);
}

// ------------- attn[b,c] = sigmoid( sum_n softmax_n(q)[n,c] * kp[b,h,n] ) * T
__global__ __launch_bounds__(256) void attn_kernel(
    const float* __restrict__ q, const float* __restrict__ mxq,
    const float* __restrict__ seq, const float* __restrict__ kp,
    const float* __restrict__ temp, float* __restrict__ attn)
{
    const int lane = threadIdx.x & 31;
    const int wid  = threadIdx.x >> 5;
    const int col  = blockIdx.x * 8 + wid;     // (b, c)
    const int b = col / DIMC, c = col % DIMC;
    const int hh = c / CHD;
    const float m = mxq[col], inv = 1.f / seq[col];
    const float* kprow = kp + (size_t)(b * HEADS + hh) * NTOK;

    float acc = 0.f;
    for (int n = lane; n < NTOK; n += 32)
        acc += __expf(q[((size_t)b * NTOK + n) * DIMC + c] - m) * inv * kprow[n];
    acc = wred_sum(acc);
    if (lane == 0)
        attn[col] = temp[hh] / (1.f + __expf(-acc));
}

// ---------------------------- depthwise 3x3 + bias, fused gamma*(.)+residual
// block = (channel chunk of 256, image row y, batch b); bf16 in/out.
#define CB 256
__global__ __launch_bounds__(256) void dwconv_kernel(
    const unsigned short* __restrict__ h1, const float* __restrict__ w,
    const float* __restrict__ dwb, const float* __restrict__ gamma,
    unsigned short* __restrict__ h2)
{
    __shared__ unsigned short tile[3][WSZ][CB];   // 48 KB
    const int c0 = blockIdx.x * CB;
    const int y  = blockIdx.y;
    const int b  = blockIdx.z;

    #pragma unroll
    for (int r = 0; r < 3; ++r) {
        const int yy = y + r - 1;
        for (int i = threadIdx.x; i < (WSZ * CB) / 8; i += 256) {
            const int x  = (i * 8) / CB;
            const int cc = (i * 8) % CB;
            uint4 v = make_uint4(0u, 0u, 0u, 0u);
            if (yy >= 0 && yy < HS)
                v = *(const uint4*)&h1[(((size_t)b * NTOK) + yy * WSZ + x) * HIDDEN + c0 + cc];
            *(uint4*)&tile[r][x][cc] = v;
        }
    }
    __syncthreads();

    for (int i = threadIdx.x; i < WSZ * CB; i += 256) {
        const int x  = i / CB;
        const int cc = i % CB;
        const int c  = c0 + cc;
        const float* wp = w + (size_t)c * 9;
        float a = dwb[c];
        #pragma unroll
        for (int dy = 0; dy < 3; ++dy) {
            #pragma unroll
            for (int dx = 0; dx < 3; ++dx) {
                const int xx = x + dx - 1;
                if (xx >= 0 && xx < WSZ)
                    a += wp[dy * 3 + dx] * bf2f(tile[dy][xx][cc]);
            }
        }
        const float center = bf2f(tile[1][x][cc]);
        h2[(((size_t)b * NTOK) + y * WSZ + x) * HIDDEN + c] = f2bf(gamma[c] * a + center);
    }
}

// ----------------- fc2 bias + LayerNorm + attention scale -> both outputs
__global__ __launch_bounds__(256) void ln_out_kernel(
    const float* __restrict__ yv, const float* __restrict__ fc2b,
    const float* __restrict__ lnw, const float* __restrict__ lnb,
    const float* __restrict__ attn,
    float* __restrict__ out1, float* __restrict__ out2)
{
    __shared__ float rowv[DIMC];
    __shared__ float sred[8];
    __shared__ float s_mu, s_rstd;
    const int lane = threadIdx.x & 31;
    const int wid  = threadIdx.x >> 5;
    const int bn = blockIdx.x;
    const int b = bn >> 10, n = bn & 1023;

    float s = 0.f;
    for (int c = threadIdx.x; c < DIMC; c += 256) {
        const float v = yv[(size_t)bn * DIMC + c] + fc2b[c];
        rowv[c] = v;
        s += v;
    }
    s = wred_sum(s);
    if (lane == 0) sred[wid] = s;
    __syncthreads();
    if (threadIdx.x == 0) {
        float t = 0.f;
        #pragma unroll
        for (int wv = 0; wv < 8; ++wv) t += sred[wv];
        s_mu = t * (1.f / DIMC);
    }
    __syncthreads();
    const float mu = s_mu;
    float vs = 0.f;
    for (int c = threadIdx.x; c < DIMC; c += 256) {
        const float d = rowv[c] - mu;
        vs += d * d;
    }
    vs = wred_sum(vs);
    if (lane == 0) sred[wid] = vs;
    __syncthreads();
    if (threadIdx.x == 0) {
        float t = 0.f;
        #pragma unroll
        for (int wv = 0; wv < 8; ++wv) t += sred[wv];
        s_rstd = rsqrtf(t * (1.f / DIMC) + LN_EPS);
    }
    __syncthreads();
    const float rstd = s_rstd;
    for (int c = threadIdx.x; c < DIMC; c += 256) {
        const float v = (rowv[c] - mu) * rstd * lnw[c] + lnb[c];
        const float p = attn[b * DIMC + c] * v;
        // outputs are never re-read: stream past the caches (NT stores)
        __builtin_nontemporal_store(p, &out1[(size_t)bn * DIMC + c]);
        const int hh = c / CHD, ch = c % CHD;
        __builtin_nontemporal_store(
            p, &out2[(((size_t)(b * HEADS + hh)) * NTOK + n) * CHD + ch]);
    }
}

// ================================================================== launcher
extern "C" void kernel_launch(void* const* d_in, const int* in_sizes, int n_in,
                              void* d_out, int out_size, void* d_ws, size_t ws_size,
                              hipStream_t stream)
{
    (void)in_sizes; (void)n_in; (void)out_size; (void)ws_size;
    const float* x     = (const float*)d_in[0];
    const float* Wq    = (const float*)d_in[1];
    const float* temp  = (const float*)d_in[2];
    const float* fc1w  = (const float*)d_in[3];
    const float* fc1b  = (const float*)d_in[4];
    const float* dww   = (const float*)d_in[5];
    const float* dwb   = (const float*)d_in[6];
    const float* gamma = (const float*)d_in[7];
    const float* fc2w  = (const float*)d_in[8];
    const float* fc2b  = (const float*)d_in[9];
    const float* lnw   = (const float*)d_in[10];
    const float* lnb   = (const float*)d_in[11];

    const int M = BB * NTOK;               // 16384 rows
    char* ws = (char*)d_ws;
    size_t off = 0;
    auto alloc = [&](size_t bytes) {
        size_t p = off; off += (bytes + 255) & ~(size_t)255; return p;
    };
    unsigned short* xb    = (unsigned short*)(ws + alloc((size_t)M * DIMC * 2));
    unsigned short* wqb   = (unsigned short*)(ws + alloc((size_t)DIMC * DIMC * 2));
    unsigned short* fc1wb = (unsigned short*)(ws + alloc((size_t)HIDDEN * DIMC * 2));
    unsigned short* fc2wb = (unsigned short*)(ws + alloc((size_t)DIMC * HIDDEN * 2));
    float*          qf    = (float*)(ws + alloc((size_t)M * DIMC * 4)); // reused as y
    unsigned short* h1b   = (unsigned short*)(ws + alloc((size_t)M * HIDDEN * 2));
    unsigned short* h2b   = (unsigned short*)(ws + alloc((size_t)M * HIDDEN * 2));
    float* mxx  = (float*)(ws + alloc((size_t)BB * DIMC * 4));
    float* sex  = (float*)(ws + alloc((size_t)BB * DIMC * 4));
    float* mxq  = (float*)(ws + alloc((size_t)BB * DIMC * 4));
    float* seq  = (float*)(ws + alloc((size_t)BB * DIMC * 4));
    float* kp   = (float*)(ws + alloc((size_t)BB * HEADS * NTOK * 4));
    float* attn = (float*)(ws + alloc((size_t)BB * DIMC * 4));

    float* out1 = (float*)d_out;
    float* out2 = out1 + (size_t)M * DIMC;

    // 1) bf16 conversions
    {
        int n = M * DIMC;
        cvt_f32_bf16<<<(n + 255) / 256, 256, 0, stream>>>(x, xb, n);
        n = DIMC * DIMC;
        cvt_f32_bf16<<<(n + 255) / 256, 256, 0, stream>>>(Wq, wqb, n);
        n = HIDDEN * DIMC;
        cvt_f32_bf16<<<(n + 255) / 256, 256, 0, stream>>>(fc1w, fc1wb, n);
        n = DIMC * HIDDEN;
        cvt_f32_bf16<<<(n + 255) / 256, 256, 0, stream>>>(fc2w, fc2wb, n);
    }
    // 2) q = x @ Wq^T  (f32 out)
    gemm_bf16_wmma<<<dim3(DIMC / 128, M / 128), 256, 0, stream>>>(
        xb, wqb, nullptr, qf, nullptr, M, DIMC, DIMC);
    // 3) softmax stats over tokens for x and q
    col_stats<<<(BB * DIMC) / 8, 256, 0, stream>>>(x,  mxx, sex, NTOK, DIMC);
    col_stats<<<(BB * DIMC) / 8, 256, 0, stream>>>(qf, mxq, seq, NTOK, DIMC);
    // 4) kp then attention gate
    kp_kernel<<<M, 256, 0, stream>>>(x, mxx, sex, kp);
    attn_kernel<<<(BB * DIMC) / 8, 256, 0, stream>>>(qf, mxq, seq, kp, temp, attn);
    // 5) h1 = x @ fc1^T + b  (bf16 out, L2-resident)
    gemm_bf16_wmma<<<dim3(HIDDEN / 128, M / 128), 256, 0, stream>>>(
        xb, fc1wb, fc1b, nullptr, h1b, M, HIDDEN, DIMC);
    // 6) h2 = gamma * dwconv3x3(h1) + h1  (bf16 out)
    dwconv_kernel<<<dim3(HIDDEN / CB, HS, BB), 256, 0, stream>>>(
        h1b, dww, dwb, gamma, h2b);
    // 7) y = h2 @ fc2^T  (f32 out; reuses q buffer)
    gemm_bf16_wmma<<<dim3(DIMC / 128, M / 128), 256, 0, stream>>>(
        h2b, fc2wb, nullptr, qf, nullptr, M, DIMC, HIDDEN);
    // 8) bias + LN + attention scale -> both tuple outputs
    ln_out_kernel<<<M, 256, 0, stream>>>(qf, fc2b, lnw, lnb, attn, out1, out2);
}
